// MultiGraphConvolutionLayers_5884105195571
// MI455X (gfx1250) — matrix operand order repaired and checked by reading, not compiled, not run on hardware
//
#include <hip/hip_runtime.h>
#include <hip/hip_bf16.h>

typedef __attribute__((ext_vector_type(16))) __bf16 v16bf;
typedef __attribute__((ext_vector_type(8)))  float  v8f;

#define BB 32
#define NN 1024
#define FF 32
#define EE 4
#define UU 32

// ---------------------------------------------------------------------------
// P kernel: per-edge linear  HpreT[b,e,u,m] = (ann[b,m,:] @ Wa[e] + ba[e])  (bf16, transposed)
//           residual linear  R[b,n,u]      = (ann[b,n,:] @ W2    + b2)     (f32, natural)
// blockIdx.y in [0, EE]: y < EE -> edge-type path, y == EE -> residual path.
// ann = src0 (32 features) [+ src1 (32 features) when F == 64, the concat layer].
// ---------------------------------------------------------------------------
template<int F>
__global__ __launch_bounds__(128)
void gcn_pre_kernel(const float* __restrict__ src0,
                    const float* __restrict__ src1,
                    const float* __restrict__ Wa,
                    const float* __restrict__ ba,
                    const float* __restrict__ W2,
                    const float* __restrict__ b2,
                    __bf16* __restrict__ hpreT,
                    float*  __restrict__ R)
{
  const int m = blockIdx.x * 128 + threadIdx.x;
  const int e = blockIdx.y;           // 0..EE (EE == residual)
  const int b = blockIdx.z;

  const float* W;
  const float* bias;
  if (e < EE) { W = Wa + (size_t)e * F * UU; bias = ba + (size_t)e * UU; }
  else        { W = W2;                       bias = b2; }

  float acc[UU];
#pragma unroll
  for (int u = 0; u < UU; ++u) acc[u] = bias[u];

  const float* r0 = src0 + ((size_t)b * NN + m) * FF;
  for (int f = 0; f < FF; ++f) {
    const float av = r0[f];
    const float* wr = W + (size_t)f * UU;
#pragma unroll
    for (int u = 0; u < UU; ++u) acc[u] = fmaf(av, wr[u], acc[u]);
  }
  if (F > FF) {
    const float* r1 = src1 + ((size_t)b * NN + m) * FF;
    for (int f = 0; f < FF; ++f) {
      const float av = r1[f];
      const float* wr = W + (size_t)(FF + f) * UU;
#pragma unroll
      for (int u = 0; u < UU; ++u) acc[u] = fmaf(av, wr[u], acc[u]);
    }
  }

  if (e < EE) {
    __bf16* o = hpreT + ((size_t)(b * EE + e) * UU) * NN + m;
#pragma unroll
    for (int u = 0; u < UU; ++u) o[(size_t)u * NN] = (__bf16)acc[u];   // coalesced over m
  } else {
    float* o = R + ((size_t)b * NN + m) * UU;                          // natural [b,n,u]
#pragma unroll
    for (int u = 0; u < UU; ++u) o[u] = acc[u];                        // 128B/lane vector stores
  }
}

// ---------------------------------------------------------------------------
// M kernel: part[epair,b,n,u] = sum_{e in pair} adj[b,e,n,:] @ HpreT[b,e]^T   (raw f32)
// blockIdx.y selects the edge pair -> 2x the independent waves for MLP.
// 128 threads = 4 waves; each wave owns a 16-row tile, 2 u-tiles (U=32),
// K loop in steps of 32 using v_wmma_f32_16x16x32_bf16 (f32 accumulate).
// ---------------------------------------------------------------------------
__global__ __launch_bounds__(128)
void gcn_adjmm_kernel(const float*  __restrict__ adj,     // [B,E,N,N] f32
                      const __bf16* __restrict__ hpreT,   // [B,E,U,N] bf16
                      float*        __restrict__ part)    // [2,B,N,U] f32
{
  const int tid    = threadIdx.x;
  const int wv     = tid >> 5;
  const int lane   = tid & 31;
  const int lane15 = lane & 15;
  const int khalf  = lane >> 4;

  const int b     = blockIdx.z;
  const int epair = blockIdx.y;
  const int nbase = blockIdx.x * 64 + wv * 16;

  v8f accs[2];
#pragma unroll
  for (int i = 0; i < 8; ++i) { accs[0][i] = 0.f; accs[1][i] = 0.f; }

  const int arow = nbase + lane15;     // A-frag: row M = lane&15
  const int aoff = khalf * 16;         // A-frag: K half per lane group (matches B)
  const int koff = khalf * 16;         // B-frag: K half per lane group

#pragma unroll
  for (int ei = 0; ei < 2; ++ei) {
    const int e = epair * 2 + ei;
    const float*  ap  = adj   + ((size_t)(b * EE + e) * NN + arow) * NN;
    const __bf16* b0p = hpreT + ((size_t)(b * EE + e) * UU + lane15) * NN + koff;
    const __bf16* b1p = b0p + (size_t)16 * NN;

#pragma unroll 4
    for (int kc = 0; kc < NN; kc += 32) {
      // A fragment: lanes 0-15 hold K[kc+0..15]; lanes 16-31 K[kc+16..31]
      const float4 s0 = *(const float4*)(ap + kc + aoff);
      const float4 s1 = *(const float4*)(ap + kc + aoff + 4);
      const float4 s2 = *(const float4*)(ap + kc + aoff + 8);
      const float4 s3 = *(const float4*)(ap + kc + aoff + 12);
      v16bf af;
      af[0]  = (__bf16)s0.x; af[1]  = (__bf16)s0.y; af[2]  = (__bf16)s0.z; af[3]  = (__bf16)s0.w;
      af[4]  = (__bf16)s1.x; af[5]  = (__bf16)s1.y; af[6]  = (__bf16)s1.z; af[7]  = (__bf16)s1.w;
      af[8]  = (__bf16)s2.x; af[9]  = (__bf16)s2.y; af[10] = (__bf16)s2.z; af[11] = (__bf16)s2.w;
      af[12] = (__bf16)s3.x; af[13] = (__bf16)s3.y; af[14] = (__bf16)s3.z; af[15] = (__bf16)s3.w;

      // B fragments: lane holds column u, 16 consecutive K values -> one 32B contiguous load
      const v16bf bf0 = *(const v16bf*)(b0p + kc);
      const v16bf bf1 = *(const v16bf*)(b1p + kc);

      accs[0] = __builtin_amdgcn_wmma_f32_16x16x32_bf16(false, af, false, bf0,
                                                        (short)0, accs[0], false, false);
      accs[1] = __builtin_amdgcn_wmma_f32_16x16x32_bf16(false, af, false, bf1,
                                                        (short)0, accs[1], false, false);
    }
  }

  // Store raw partials. C layout: lane col u = utile*16 + (lane&15);
  // VGPR r -> row n = nbase + r + (lane>=16 ? 8 : 0)
  const int n0 = nbase + khalf * 8;
#pragma unroll
  for (int t = 0; t < 2; ++t) {
    const int u = t * 16 + lane15;
    float* op = part + (((size_t)epair * BB + b) * NN + n0) * UU + u;
#pragma unroll
    for (int r = 0; r < 8; ++r)
      op[(size_t)r * UU] = accs[t][r];
  }
}

// ---------------------------------------------------------------------------
// Combine: out[b,n,u] = tanh(part[0,b,n,u] + part[1,b,n,u] + R[b,n,u])
// Fully coalesced, 16MB total traffic.
// ---------------------------------------------------------------------------
__global__ __launch_bounds__(256)
void gcn_combine_kernel(const float* __restrict__ part,
                        const float* __restrict__ R,
                        float*       __restrict__ out)
{
  const size_t i = (size_t)blockIdx.x * 256 + threadIdx.x;   // over B*N*U
  const float v = part[i] + part[(size_t)BB * NN * UU + i] + R[i];
  out[i] = tanhf(v);
}

// ---------------------------------------------------------------------------
// Host launcher: P0 -> M0 -> C0 (h1) -> P1 (concat) -> M1 -> C1 (d_out)
// Workspace: [0,8M) HpreT bf16 | [8M,12M) R f32 | [12M,16M) h1 f32 | [16M,24M) part f32
// ---------------------------------------------------------------------------
extern "C" void kernel_launch(void* const* d_in, const int* in_sizes, int n_in,
                              void* d_out, int out_size, void* d_ws, size_t ws_size,
                              hipStream_t stream) {
  const float* n_tensor = (const float*)d_in[0];
  const float* adj      = (const float*)d_in[1];
  const float* W_adj0   = (const float*)d_in[2];
  const float* b_adj0   = (const float*)d_in[3];
  const float* W2_0     = (const float*)d_in[4];
  const float* b2_0     = (const float*)d_in[5];
  const float* W_adj1   = (const float*)d_in[6];
  const float* b_adj1   = (const float*)d_in[7];
  const float* W2_1     = (const float*)d_in[8];
  const float* b2_1     = (const float*)d_in[9];
  float* out = (float*)d_out;

  char*   ws    = (char*)d_ws;
  __bf16* hpreT = (__bf16*)ws;                              // 8 MB
  float*  R     = (float*)(ws + (size_t)8  * 1024 * 1024);  // 4 MB
  float*  h1    = (float*)(ws + (size_t)12 * 1024 * 1024);  // 4 MB
  float*  part  = (float*)(ws + (size_t)16 * 1024 * 1024);  // 8 MB

  dim3 pgrid(NN / 128, EE + 1, BB), pblk(128);
  dim3 mgrid(NN / 64, 2, BB),       mblk(128);
  dim3 cgrid((BB * NN * UU) / 256), cblk(256);

  // Layer 0
  gcn_pre_kernel<32><<<pgrid, pblk, 0, stream>>>(n_tensor, nullptr,
                                                 W_adj0, b_adj0, W2_0, b2_0, hpreT, R);
  gcn_adjmm_kernel<<<mgrid, mblk, 0, stream>>>(adj, hpreT, part);
  gcn_combine_kernel<<<cgrid, cblk, 0, stream>>>(part, R, h1);

  // Layer 1 (annotations = concat(n_tensor, h1), F = 64)
  gcn_pre_kernel<64><<<pgrid, pblk, 0, stream>>>(n_tensor, h1,
                                                 W_adj1, b_adj1, W2_1, b2_1, hpreT, R);
  gcn_adjmm_kernel<<<mgrid, mblk, 0, stream>>>(adj, hpreT, part);
  gcn_combine_kernel<<<cgrid, cblk, 0, stream>>>(part, R, out);
}
